// GraphAttentionLayer_27693949124679
// MI455X (gfx1250) — matrix operand order repaired and checked by reading, not compiled, not run on hardware
//
#include <hip/hip_runtime.h>
#include <hip/hip_bf16.h>

#define N_NODES 4096
#define FIN     512
#define FOUT    128
#define NHEAD   4
#define ALPHA   0.2f

typedef __bf16 bf16_t;
typedef __attribute__((ext_vector_type(16))) __bf16 v16bf;
typedef __attribute__((ext_vector_type(8)))  __bf16 v8bf;
typedef __attribute__((ext_vector_type(8)))  float  v8f;
typedef __attribute__((ext_vector_type(4)))  float  v4f;
typedef __attribute__((ext_vector_type(4)))  int    v4i;

#define AS1 __attribute__((address_space(1)))
#define AS3 __attribute__((address_space(3)))

// --- CDNA5 async global->LDS staging (ASYNCcnt path), with safe fallback ----
#if __has_builtin(__builtin_amdgcn_global_load_async_to_lds_b128)
#define HAVE_ASYNC_LDS 1
#endif

__device__ __forceinline__ void async_copy16(const void* g, void* l) {
#if defined(HAVE_ASYNC_LDS)
    // Builtin signature (from hipcc diagnostic): (v4i AS1*, v4i AS3*, imm, imm)
    __builtin_amdgcn_global_load_async_to_lds_b128(
        (AS1 v4i*)g, (AS3 v4i*)l, /*offset=*/0, /*cpol=*/0);
#else
    *(v4i*)l = *(const v4i*)g;
#endif
}

__device__ __forceinline__ void wait_async_all() {
#if defined(HAVE_ASYNC_LDS)
# if __has_builtin(__builtin_amdgcn_s_wait_asynccnt)
    __builtin_amdgcn_s_wait_asynccnt(0);
# else
    asm volatile("s_wait_asynccnt 0x0" ::: "memory");
# endif
#endif
}

// ---------------------------------------------------------------------------
// Kernel 0: pack W into fragment-native bf16 B-layout:
//   Wp[((k*16 + c)*FOUT + o)*32 + kk] = bf16(W[k][c*32 + kk][o])
// so a kernel-1 lane's B-fragment is two contiguous b128 loads.
// ---------------------------------------------------------------------------
__global__ __launch_bounds__(256) void gat_pack_w(
    const float* __restrict__ W, bf16_t* __restrict__ Wp)
{
    const int gid = blockIdx.x * 256 + threadIdx.x;   // over K*FIN*FOUT
    const int k   = gid >> 16;                        // FIN*FOUT = 65536
    const int rem = gid & 65535;
    const int f   = rem >> 7;
    const int o   = rem & (FOUT - 1);
    const int c   = f >> 5;
    const int kk  = f & 31;
    Wp[(((size_t)(k * 16 + c) * FOUT) + o) * 32 + kk] = (bf16_t)W[gid];
}

// ---------------------------------------------------------------------------
// Kernel 1: H[k][n][o] = sum_f x[n][f] * W[k][f][o]   (bf16 WMMA, f32 accum)
// One wave per (16-node tile, head). x tile async-staged in LDS, shared by
// the 4 head-waves. Writes H (f32, [k][n][o]) and Ht (bf16, [k][o][m]).
// ---------------------------------------------------------------------------
__global__ __launch_bounds__(128) void gat_h_gemm(
    const float* __restrict__ x, const bf16_t* __restrict__ Wp,
    float* __restrict__ H, bf16_t* __restrict__ Ht)
{
    __shared__ float xs[16][32];
    const int lane  = threadIdx.x;        // 0..31
    const int head  = threadIdx.y;        // 0..3
    const int nbase = blockIdx.x * 16;
    const int row   = lane & 15;
    const int hi    = lane >> 4;

    v8f acc[8];
#pragma unroll
    for (int t = 0; t < 8; ++t)
#pragma unroll
        for (int r = 0; r < 8; ++r) acc[t][r] = 0.0f;

    for (int c = 0; c < FIN; c += 32) {
        const int ci = c >> 5;
        // cooperative async stage of x[16 rows][32 f] (128 threads x b128)
        const int t4   = (threadIdx.y * 32 + threadIdx.x) * 4;
        const int srow = t4 >> 5, scol = t4 & 31;
        async_copy16(&x[(size_t)(nbase + srow) * FIN + c + scol], &xs[srow][scol]);
        wait_async_all();
        __syncthreads();

        // A fragment: 16x32 bf16, 16-bit A layout (lane half selects K octet)
        v16bf afrag;
#pragma unroll
        for (int j = 0; j < 8; ++j) {
            afrag[j]     = (bf16_t)xs[row][hi * 8 + j];        // K = hi*8 + j
            afrag[8 + j] = (bf16_t)xs[row][16 + hi * 8 + j];   // K = 16 + hi*8 + j
        }

#pragma unroll
        for (int t = 0; t < 8; ++t) {
            const int o = t * 16 + row;   // B column = lane&15
            const bf16_t* src =
                Wp + (((size_t)(head * 16 + ci) * FOUT) + o) * 32 + hi * 8;
            v8bf b0 = *(const v8bf*)src;        // K = hi*8 + 0..7
            v8bf b1 = *(const v8bf*)(src + 16); // K = 16 + hi*8 + 0..7
            v16bf bfrag;
#pragma unroll
            for (int e = 0; e < 8; ++e) { bfrag[e] = b0[e]; bfrag[8 + e] = b1[e]; }
            acc[t] = __builtin_amdgcn_wmma_f32_16x16x32_bf16(
                false, afrag, false, bfrag, (short)0, acc[t], false, false);
        }
        __syncthreads();
    }

    // Store: f32 H[k][n][o] and bf16 transposed Ht[k][o][n] (packed b128 store)
#pragma unroll
    for (int t = 0; t < 8; ++t) {
        const int o = t * 16 + row;
        v8bf pk;
#pragma unroll
        for (int r = 0; r < 8; ++r) {
            const int n = nbase + r + 8 * hi;   // C layout: row = r (+8 upper half)
            H[((size_t)head * N_NODES + n) * FOUT + o] = acc[t][r];
            pk[r] = (bf16_t)acc[t][r];
        }
        *(v8bf*)&Ht[((size_t)(head * FOUT + o)) * N_NODES + nbase + 8 * hi] = pk;
    }
}

// ---------------------------------------------------------------------------
// Kernel 2: e_i[k][n] = H[k][n][:] . a_src[k]; e_j[k][n] = H[k][n][:] . a_dst[k]
// ---------------------------------------------------------------------------
__global__ __launch_bounds__(256) void gat_edge_scores(
    const float* __restrict__ H, const float* __restrict__ a,
    float* __restrict__ ei, float* __restrict__ ej)
{
    const int gid = blockIdx.x * 256 + threadIdx.x;
    const int k = gid >> 12;            // 4096 nodes per head
    const int n = gid & (N_NODES - 1);
    const float* h    = H + ((size_t)k * N_NODES + n) * FOUT;
    const float* asrc = a + (size_t)k * 2 * FOUT;
    const float* adst = asrc + FOUT;
    float s0 = 0.f, s1 = 0.f;
    for (int o = 0; o < FOUT; o += 4) {
        v4f hv = *(const v4f*)&h[o];
        v4f av = *(const v4f*)&asrc[o];
        v4f dv = *(const v4f*)&adst[o];
        s0 += hv.x * av.x + hv.y * av.y + hv.z * av.z + hv.w * av.w;
        s1 += hv.x * dv.x + hv.y * dv.y + hv.z * dv.z + hv.w * dv.w;
    }
    ei[(size_t)k * N_NODES + n] = s0;
    ej[(size_t)k * N_NODES + n] = s1;
}

// ---------------------------------------------------------------------------
// Kernel 3: per-row online softmax stats (max, sum) over masked neighbors,
// all 4 heads per pass over the adj row. One wave32 per row, shuffle-reduce.
// ---------------------------------------------------------------------------
__global__ __launch_bounds__(128) void gat_softmax_stats(
    const int* __restrict__ adj, const float* __restrict__ ei,
    const float* __restrict__ ej, float* __restrict__ mx, float* __restrict__ sm)
{
    const int lane = threadIdx.x;
    const int n    = blockIdx.x * 4 + threadIdx.y;

    float eiv[NHEAD], m_[NHEAD], s_[NHEAD];
#pragma unroll
    for (int k = 0; k < NHEAD; ++k) {
        eiv[k] = ei[(size_t)k * N_NODES + n];
        m_[k] = -1e30f;                 // finite sentinel: avoids inf-inf NaN
        s_[k] = 0.f;
    }

    const int* arow = adj + (size_t)n * N_NODES;
    for (int mcol = lane; mcol < N_NODES; mcol += 32) {
        if (arow[mcol] != 0) {
#pragma unroll
            for (int k = 0; k < NHEAD; ++k) {
                float s = eiv[k] + ej[(size_t)k * N_NODES + mcol];
                s = s > 0.f ? s : ALPHA * s;                  // leaky relu
                if (s > m_[k]) { s_[k] = s_[k] * __expf(m_[k] - s) + 1.f; m_[k] = s; }
                else           { s_[k] += __expf(s - m_[k]); }
            }
        }
    }

#pragma unroll
    for (int k = 0; k < NHEAD; ++k) {
        float mk = m_[k], sk = s_[k];
#pragma unroll
        for (int off = 16; off > 0; off >>= 1) {
            float om = __shfl_xor(mk, off, 32);
            float os = __shfl_xor(sk, off, 32);
            float M  = fmaxf(mk, om);
            sk = sk * __expf(mk - M) + os * __expf(om - M);
            mk = M;
        }
        if (lane == 0) {
            mx[(size_t)k * N_NODES + n] = mk;
            sm[(size_t)k * N_NODES + n] = sk;
        }
    }
}

// ---------------------------------------------------------------------------
// Kernel 4: out[n][k*128+o] = (1/sum_row) * sum_m exp_p(n,k,m) * H[m][k][o]
// Fused: unnormalized P built on the fly in A-fragment layout (bf16), Ht gives
// contiguous B-fragments, f32 WMMA accumulation; 1/sum applied in epilogue.
// adj tile async-staged in LDS once per block, shared by the 4 head-waves.
// ---------------------------------------------------------------------------
__global__ __launch_bounds__(128) void gat_aggregate(
    const int* __restrict__ adj, const bf16_t* __restrict__ Ht,
    const float* __restrict__ ei, const float* __restrict__ ej,
    const float* __restrict__ mx, const float* __restrict__ sm,
    float* __restrict__ out)
{
    __shared__ int adjs[16][32];
    const int lane  = threadIdx.x;
    const int head  = threadIdx.y;
    const int nbase = blockIdx.x * 16;
    const int row   = lane & 15;
    const int hi    = lane >> 4;
    const int n     = nbase + row;

    const float eiv  = ei[(size_t)head * N_NODES + n];
    const float mrow = mx[(size_t)head * N_NODES + n];
    const float* ejk = ej + (size_t)head * N_NODES;

    v8f acc[8];
#pragma unroll
    for (int t = 0; t < 8; ++t)
#pragma unroll
        for (int r = 0; r < 8; ++r) acc[t][r] = 0.0f;

    for (int mb = 0; mb < N_NODES; mb += 32) {
        // cooperative async stage of adj[16 rows][32 m]
        const int t4   = (threadIdx.y * 32 + threadIdx.x) * 4;
        const int srow = t4 >> 5, scol = t4 & 31;
        async_copy16(&adj[(size_t)(nbase + srow) * N_NODES + mb + scol],
                     &adjs[srow][scol]);
        if (mb + 32 < N_NODES)   // prefetch next adj tile -> global_prefetch_b8
            __builtin_prefetch(&adj[(size_t)(nbase + srow) * N_NODES + mb + 32 + scol], 0, 3);
        wait_async_all();
        __syncthreads();

        // A fragment: unnormalized softmax numerators, 16x32 bf16 A layout
        v16bf afrag;
#pragma unroll
        for (int j = 0; j < 8; ++j) {
            {
                const int ml = hi * 8 + j;
                float s = eiv + ejk[mb + ml];
                s = s > 0.f ? s : ALPHA * s;
                afrag[j] = (bf16_t)(adjs[row][ml] ? __expf(s - mrow) : 0.f);
            }
            {
                const int ml = 16 + hi * 8 + j;
                float s = eiv + ejk[mb + ml];
                s = s > 0.f ? s : ALPHA * s;
                afrag[8 + j] = (bf16_t)(adjs[row][ml] ? __expf(s - mrow) : 0.f);
            }
        }

#pragma unroll
        for (int t = 0; t < 8; ++t) {
            const int o = t * 16 + row;  // B column index = lane&15
            const bf16_t* src =
                Ht + ((size_t)(head * FOUT + o)) * N_NODES + mb + hi * 8;
            v8bf b0 = *(const v8bf*)src;        // K = hi*8 + 0..7
            v8bf b1 = *(const v8bf*)(src + 16); // K = 16 + hi*8 + 0..7
            v16bf bfrag;
#pragma unroll
            for (int e = 0; e < 8; ++e) { bfrag[e] = b0[e]; bfrag[8 + e] = b1[e]; }
            acc[t] = __builtin_amdgcn_wmma_f32_16x16x32_bf16(
                false, afrag, false, bfrag, (short)0, acc[t], false, false);
        }
        __syncthreads();
    }

    // Epilogue: apply per-row 1/sum (C layout row = r + 8*hi), store f32 out.
    float isv[8];
#pragma unroll
    for (int r = 0; r < 8; ++r)
        isv[r] = 1.0f / sm[(size_t)head * N_NODES + nbase + r + 8 * hi];

#pragma unroll
    for (int t = 0; t < 8; ++t) {
        const int o = t * 16 + row;
#pragma unroll
        for (int r = 0; r < 8; ++r) {
            const int nn = nbase + r + 8 * hi;
            out[(size_t)nn * (NHEAD * FOUT) + head * FOUT + o] = acc[t][r] * isv[r];
        }
    }
}

// ---------------------------------------------------------------------------
extern "C" void kernel_launch(void* const* d_in, const int* in_sizes, int n_in,
                              void* d_out, int out_size, void* d_ws, size_t ws_size,
                              hipStream_t stream)
{
    (void)in_sizes; (void)n_in; (void)out_size; (void)ws_size;

    const float* x   = (const float*)d_in[0];   // (4096, 512)  f32
    const int*   adj = (const int*)  d_in[1];   // (4096, 4096) i32
    const float* W   = (const float*)d_in[2];   // (4, 512, 128) f32
    const float* a   = (const float*)d_in[3];   // (4, 256, 1)  f32
    float*       out = (float*)d_out;           // (4096, 512)  f32

    // Workspace carving (12.75 MB total)
    char* ws = (char*)d_ws;
    float*  H  = (float*)ws;                                             // 8 MB
    size_t  off = (size_t)NHEAD * N_NODES * FOUT * sizeof(float);
    bf16_t* Ht = (bf16_t*)(ws + off);                                    // 4 MB
    off += (size_t)NHEAD * N_NODES * FOUT * sizeof(bf16_t);
    float* ei = (float*)(ws + off); off += (size_t)NHEAD * N_NODES * 4;  // 64 KB
    float* ej = (float*)(ws + off); off += (size_t)NHEAD * N_NODES * 4;  // 64 KB
    float* mx = (float*)(ws + off); off += (size_t)NHEAD * N_NODES * 4;  // 64 KB
    float* sm = (float*)(ws + off); off += (size_t)NHEAD * N_NODES * 4;  // 64 KB
    bf16_t* Wp = (bf16_t*)(ws + off);                                    // 512 KB

    dim3 blk(32, 4);  // 4 wave32s: one head per wave
    gat_pack_w       <<<(NHEAD * FIN * FOUT) / 256, 256, 0, stream>>>(W, Wp);
    gat_h_gemm       <<<N_NODES / 16, blk, 0, stream>>>(x, Wp, H, Ht);
    gat_edge_scores  <<<(N_NODES * NHEAD) / 256, 256, 0, stream>>>(H, a, ei, ej);
    gat_softmax_stats<<<N_NODES / 4, blk, 0, stream>>>(adj, ei, ej, mx, sm);
    gat_aggregate    <<<N_NODES / 16, blk, 0, stream>>>(adj, Ht, ei, ej, mx, sm, out);
}
